// GraphKMeans_27917287424729
// MI455X (gfx1250) — compile-verified, matrix operand order checked by reading, not run on hardware
//
#include <hip/hip_runtime.h>

#define N_POINTS   131072
#define D_EMBED    512
#define K_CLUSTERS 512
#define ROWS_PER_BLOCK 128
#define THREADS    256
#define PITCH      520                     // bf16 elems per LDS row (512 + 8 pad -> conflict-free b128 lanes)
#define NBLOCKS    (N_POINTS / ROWS_PER_BLOCK)   // 1024

typedef __attribute__((ext_vector_type(16))) __bf16 v16bf;
typedef __attribute__((ext_vector_type(8)))  float  v8f;

struct U32x8 { uint4 a, b; };

__device__ __forceinline__ unsigned short f32_to_bf16_rne(float x) {
  unsigned u = __float_as_uint(x);
  unsigned r = u + 0x7FFFu + ((u >> 16) & 1u);
  return (unsigned short)(r >> 16);
}

__device__ __forceinline__ void split_bf16(float x, unsigned short* hi, unsigned short* lo) {
  unsigned short h = f32_to_bf16_rne(x);
  *hi = h;
  float hf = __uint_as_float(((unsigned)h) << 16);
  *lo = f32_to_bf16_rne(x - hf);
}

// Load one 16-bit WMMA operand fragment: 8 bf16 at p (K+0..7) and 8 bf16 at p+16 (K+16..23).
__device__ __forceinline__ v16bf load_frag(const unsigned short* p) {
  U32x8 r;
  r.a = *reinterpret_cast<const uint4*>(p);
  r.b = *reinterpret_cast<const uint4*>(p + 16);
  return __builtin_bit_cast(v16bf, r);
}

// ---------------- prep: centroids f32 -> bf16 hi/lo + ||c||^2 ----------------
__global__ __launch_bounds__(128) void kmeans_prep(const float* __restrict__ cent,
                                                   unsigned short* __restrict__ chi,
                                                   unsigned short* __restrict__ clo,
                                                   float* __restrict__ c2) {
  __shared__ float red[128];
  const int k = blockIdx.x;
  const int t = threadIdx.x;
  float4 v = reinterpret_cast<const float4*>(cent + (size_t)k * D_EMBED)[t];
  float ss = v.x * v.x + v.y * v.y + v.z * v.z + v.w * v.w;
  unsigned short* hp = chi + (size_t)k * D_EMBED + t * 4;
  unsigned short* lp = clo + (size_t)k * D_EMBED + t * 4;
  split_bf16(v.x, hp + 0, lp + 0);
  split_bf16(v.y, hp + 1, lp + 1);
  split_bf16(v.z, hp + 2, lp + 2);
  split_bf16(v.w, hp + 3, lp + 3);
  red[t] = ss;
  __syncthreads();
  for (int off = 64; off > 0; off >>= 1) {
    if (t < off) red[t] += red[t + off];
    __syncthreads();
  }
  if (t == 0) c2[k] = red[0];
}

// ---------------- fused: split-bf16 WMMA GEMM + online softmax ----------------
__global__ __launch_bounds__(THREADS) void kmeans_fused(const float* __restrict__ emb,
                                                        const unsigned short* __restrict__ chi,
                                                        const unsigned short* __restrict__ clo,
                                                        const float* __restrict__ c2,
                                                        const float* __restrict__ alpha_p,
                                                        float* __restrict__ partials) {
  extern __shared__ char smem[];
  unsigned short* Ahi = reinterpret_cast<unsigned short*>(smem);
  unsigned short* Alo = Ahi + ROWS_PER_BLOCK * PITCH;
  float* x2s  = reinterpret_cast<float*>(Alo + ROWS_PER_BLOCK * PITCH);
  float* wsum = x2s + ROWS_PER_BLOCK;

  const int tid  = threadIdx.x;
  const int lane = tid & 31;
  const int wv   = tid >> 5;
  const int l15  = lane & 15;
  const int hf   = lane >> 4;                     // half-wave select
  const size_t row0 = (size_t)blockIdx.x * ROWS_PER_BLOCK;
  const float alpha = alpha_p[0];

  // Stage 128 rows x 512 f32 -> bf16 hi/lo in LDS (each element converted once).
  for (int idx = tid; idx < ROWS_PER_BLOCK * (D_EMBED / 4); idx += THREADS) {
    int r = idx >> 7;                             // D_EMBED/4 == 128
    int c = idx & 127;
    float4 v = reinterpret_cast<const float4*>(emb + (row0 + r) * D_EMBED)[c];
    int base = r * PITCH + c * 4;
    split_bf16(v.x, &Ahi[base + 0], &Alo[base + 0]);
    split_bf16(v.y, &Ahi[base + 1], &Alo[base + 1]);
    split_bf16(v.z, &Ahi[base + 2], &Alo[base + 2]);
    split_bf16(v.w, &Ahi[base + 3], &Alo[base + 3]);
  }
  __syncthreads();
  // ||x||^2 per row in full f32 (second read is L2-hot).
  if (tid < ROWS_PER_BLOCK) {
    const float4* rp = reinterpret_cast<const float4*>(emb + (row0 + tid) * D_EMBED);
    float a2 = 0.f;
    for (int i = 0; i < D_EMBED / 4; ++i) {
      float4 v = rp[i];
      a2 += v.x * v.x + v.y * v.y + v.z * v.z + v.w * v.w;
    }
    x2s[tid] = a2;
  }
  __syncthreads();

  const int rowBase = wv * 16;                    // this wave owns 16 rows
  float x2r[8];
#pragma unroll
  for (int i = 0; i < 8; ++i) x2r[i] = x2s[rowBase + hf * 8 + i];   // C-layout: M = i + 8*hf

  float mM[8], sS[8], tT[8];
#pragma unroll
  for (int i = 0; i < 8; ++i) { mM[i] = -3.4e38f; sS[i] = 0.f; tT[i] = 0.f; }

  // Per-lane A fragment base: row = rowBase + l15, k-chunk select by half-wave.
  const unsigned short* aHi = Ahi + (rowBase + l15) * PITCH + hf * 8;
  const unsigned short* aLo = Alo + (rowBase + l15) * PITCH + hf * 8;

  for (int j = 0; j < K_CLUSTERS / 16; ++j) {
    __syncthreads();                              // keep waves lockstep -> WGP$ reuse on B
    const int n = j * 16 + l15;                   // centroid (B column) for this lane
    const float c2l = c2[n];
    const unsigned short* bHi = chi + (size_t)n * D_EMBED + hf * 8;
    const unsigned short* bLo = clo + (size_t)n * D_EMBED + hf * 8;
    if (j + 1 < K_CLUSTERS / 16) {                // global_prefetch_b8 for the next tile
      __builtin_prefetch(chi + (size_t)(n + 16) * D_EMBED + hf * 8, 0, 1);
      __builtin_prefetch(clo + (size_t)(n + 16) * D_EMBED + hf * 8, 0, 1);
    }
    // Three independent accumulator chains (one per split-product class):
    // consecutive WMMAs never share a D->C dependency, so the XDL pipe can
    // issue back-to-back without TRANS32_DEP stalls.
    v8f acc0 = {};   // hi*hi
    v8f acc1 = {};   // hi*lo
    v8f acc2 = {};   // lo*hi
#pragma unroll 4
    for (int ks = 0; ks < D_EMBED / 32; ++ks) {
      const int kb = ks * 32;
      v16bf ah = load_frag(aHi + kb);
      v16bf al = load_frag(aLo + kb);
      v16bf bh = load_frag(bHi + kb);
      v16bf bl = load_frag(bLo + kb);
      acc0 = __builtin_amdgcn_wmma_f32_16x16x32_bf16(false, ah, false, bh, (short)0, acc0, false, false);
      acc1 = __builtin_amdgcn_wmma_f32_16x16x32_bf16(false, ah, false, bl, (short)0, acc1, false, false);
      acc2 = __builtin_amdgcn_wmma_f32_16x16x32_bf16(false, al, false, bh, (short)0, acc2, false, false);
    }
    v8f acc = (acc0 + acc1) + acc2;
    // Online softmax update over this 16-column tile (butterfly within 16-lane half-wave).
#pragma unroll
    for (int i = 0; i < 8; ++i) {
      float d  = x2r[i] - 2.0f * acc[i] + c2l;
      float lg = -alpha * d;
      float tm = lg;
      for (int msk = 1; msk < 16; msk <<= 1) tm = fmaxf(tm, __shfl_xor(tm, msk, 32));
      float e  = __expf(lg - tm);
      float se = e, te = d * e;
      for (int msk = 1; msk < 16; msk <<= 1) {
        se += __shfl_xor(se, msk, 32);
        te += __shfl_xor(te, msk, 32);
      }
      float nm  = fmaxf(mM[i], tm);
      float sc0 = __expf(mM[i] - nm);
      float sc1 = __expf(tm - nm);
      sS[i] = sS[i] * sc0 + se * sc1;
      tT[i] = tT[i] * sc0 + te * sc1;
      mM[i] = nm;
    }
  }

  // Per-wave row sum: t/s per row, 8 rows per half-wave (replicated across 16 lanes).
  float rs = 0.f;
#pragma unroll
  for (int i = 0; i < 8; ++i) rs += tT[i] / sS[i];
  rs += __shfl_xor(rs, 16, 32);                   // combine the two half-waves -> 16-row sum
  if (lane == 0) wsum[wv] = rs;
  __syncthreads();
  if (tid == 0) {
    float tot = 0.f;
    for (int w = 0; w < 8; ++w) tot += wsum[w];
    partials[blockIdx.x] = tot;                   // deterministic: no atomics
  }
}

// ---------------- final fixed-order reduction ----------------
__global__ __launch_bounds__(256) void kmeans_final(const float* __restrict__ partials,
                                                    const float* __restrict__ lambd_p,
                                                    float* __restrict__ out) {
  __shared__ float red[256];
  const int t = threadIdx.x;
  float s = 0.f;
  for (int i = t; i < NBLOCKS; i += 256) s += partials[i];
  red[t] = s;
  __syncthreads();
  for (int off = 128; off > 0; off >>= 1) {
    if (t < off) red[t] += red[t + off];
    __syncthreads();
  }
  if (t == 0) out[0] = lambd_p[0] * red[0] / (float)N_POINTS;
}

extern "C" void kernel_launch(void* const* d_in, const int* in_sizes, int n_in,
                              void* d_out, int out_size, void* d_ws, size_t ws_size,
                              hipStream_t stream) {
  (void)in_sizes; (void)n_in; (void)out_size; (void)ws_size;
  const float* emb   = (const float*)d_in[0];
  const float* cent  = (const float*)d_in[1];
  const float* alpha = (const float*)d_in[2];
  const float* lambd = (const float*)d_in[3];

  char* ws = (char*)d_ws;                         // ws layout (~1.03 MB total)
  unsigned short* chi = (unsigned short*)(ws);                 // 512*512*2 = 512 KB
  unsigned short* clo = (unsigned short*)(ws + 524288);        // 512 KB
  float* c2           = (float*)(ws + 1048576);                // 2 KB (padded to 4 KB)
  float* partials     = (float*)(ws + 1048576 + 4096);         // 4 KB

  kmeans_prep<<<K_CLUSTERS, 128, 0, stream>>>(cent, chi, clo, c2);

  const size_t shmem = (size_t)(2 * ROWS_PER_BLOCK * PITCH * sizeof(unsigned short))
                     + ROWS_PER_BLOCK * sizeof(float) + 8 * sizeof(float);   // 266,784 B
  hipFuncSetAttribute(reinterpret_cast<const void*>(kmeans_fused),
                      hipFuncAttributeMaxDynamicSharedMemorySize, (int)shmem);
  kmeans_fused<<<NBLOCKS, THREADS, shmem, stream>>>(emb, chi, clo, c2, alpha, partials);

  kmeans_final<<<1, 256, 0, stream>>>(partials, lambd, (float*)d_out);
}
